// GraphVAE_33749853012396
// MI455X (gfx1250) — matrix operand order repaired and checked by reading, not compiled, not run on hardware
//
#include <hip/hip_runtime.h>
#include <hip/hip_bf16.h>

typedef __attribute__((ext_vector_type(16))) _Float16 v16h;
typedef __attribute__((ext_vector_type(8)))  _Float16 v8h;
typedef __attribute__((ext_vector_type(4)))  _Float16 v4h;
typedef __attribute__((ext_vector_type(8)))  float    v8f;

#define GN 50000
#define GE 800000
#define FH 128
#define FEDGE 16
#define LATD 64
#define MAXND 200
#define LDSPITCH 136   // 128 + 8 f16 pad to skew LDS banks

// ---------- helpers ----------
__device__ __forceinline__ unsigned f2ord(float f) {
  unsigned u = __float_as_uint(f);
  return (u & 0x80000000u) ? ~u : (u | 0x80000000u);
}
__device__ __forceinline__ float ord2f(unsigned o) {
  unsigned u = (o & 0x80000000u) ? (o & 0x7fffffffu) : ~o;
  return __uint_as_float(u);
}

__global__ void k_fill_u32(unsigned* p, unsigned v, size_t n) {
  size_t i = (size_t)blockIdx.x * blockDim.x + threadIdx.x;
  if (i < n) p[i] = v;
}

// ---------- one-time f32 -> f16 conversion (float4 in, 4xf16 out) ----------
__global__ void k_cvt_f16(const float* __restrict__ in, _Float16* __restrict__ outp, size_t n) {
  size_t i = ((size_t)blockIdx.x * blockDim.x + threadIdx.x) * 4;
  if (i < n) {
    float4 v = *(const float4*)(in + i);
    v4h o = { (_Float16)v.x, (_Float16)v.y, (_Float16)v.z, (_Float16)v.w };
    *(v4h*)(outp + i) = o;
  }
}

// ---------- pre-swizzle B (128x128 f32) into WMMA lane layout, f16 ----------
// Bs[((kt*8 + wtile)*32 + lane)*16 + j] = B[(kt*32 + (lane>>4)*16 + j)*128 + wtile*16 + (lane&15)]
__global__ void k_prep_B(const float* __restrict__ B, _Float16* __restrict__ Bs) {
  int t = blockIdx.x * blockDim.x + threadIdx.x;   // 1024 = 4 kt * 8 wtile * 32 lane
  if (t >= 4 * 8 * 32) return;
  int lane = t & 31, wt = (t >> 5) & 7, kt = t >> 8;
  int half = lane >> 4, l16 = lane & 15;
  int n = wt * 16 + l16;
  int kbase = kt * 32 + half * 16;
  _Float16* d = Bs + (size_t)t * 16;
  #pragma unroll
  for (int j = 0; j < 16; ++j) d[j] = (_Float16)B[(size_t)(kbase + j) * FH + n];
}

// ---------- WMMA GEMM: C[M x 128] = A16[M x 128] @ Bswz, f32 accum out ----------
// block = 256 threads = 8 waves; wave w owns output columns [w*16, w*16+16)
__global__ void k_gemm128_wmma(const _Float16* __restrict__ A, const _Float16* __restrict__ Bs,
                               float* __restrict__ C, int M) {
  __shared__ _Float16 lds_a[16 * LDSPITCH];
  const int wave = threadIdx.x >> 5;
  const int lane = threadIdx.x & 31;
  const int half = lane >> 4;
  const int l16  = lane & 15;
  const int m_base = blockIdx.x * 16;
  if (m_base >= M) return;

  // cooperative A-tile stage: 256 threads x 8 f16 (global_load_b128 -> ds_store_b128)
  {
    int r = threadIdx.x >> 4, s = threadIdx.x & 15;
    *(v8h*)&lds_a[r * LDSPITCH + s * 8] = *(const v8h*)&A[(size_t)(m_base + r) * FH + s * 8];
  }
  __syncthreads();

  v8f acc = {};
  #pragma unroll
  for (int kt = 0; kt < 4; ++kt) {
    const int k0 = kt * 32;
    v8h lo = *(const v8h*)&lds_a[l16 * LDSPITCH + k0 + half * 16];
    v8h hi = *(const v8h*)&lds_a[l16 * LDSPITCH + k0 + half * 16 + 8];
    v16h a = __builtin_shufflevector(lo, hi, 0, 1, 2, 3, 4, 5, 6, 7,
                                     8, 9, 10, 11, 12, 13, 14, 15);
    v16h b = *(const v16h*)&Bs[(size_t)((kt * 8 + wave) * 32 + lane) * 16];
    acc = __builtin_amdgcn_wmma_f32_16x16x32_f16(false, a, false, b, (short)0, acc,
                                                 false, false);
  }
  const int col = wave * 16 + l16;
  #pragma unroll
  for (int j = 0; j < 8; ++j)
    C[(size_t)(m_base + half * 8 + j) * FH + col] = acc[j];
}

// ---------- degree + mean edge-attr (16 lanes per edge) ----------
__global__ void k_deg_ea(const int* __restrict__ ei, const float* __restrict__ ea,
                         float* deg, float* mea) {
  size_t t = (size_t)blockIdx.x * blockDim.x + threadIdx.x;
  size_t e = t >> 4; int f = (int)(t & 15);
  if (e >= (size_t)GE) return;
  int d = ei[(size_t)GE + e];
  if (f == 0) atomicAdd(&deg[d], 1.0f);
  atomicAdd(&mea[(size_t)d * FEDGE + f], ea[e * FEDGE + f]);
}
__global__ void k_mea_div(const float* __restrict__ deg, float* mea) {
  size_t t = (size_t)blockIdx.x * blockDim.x + threadIdx.x;
  if (t >= (size_t)GN * FEDGE) return;
  mea[t] /= fmaxf(deg[t >> 4], 1.0f);
}

// ---------- wea[f] = dot(We[f,:], att_e) ----------
__global__ void k_wea(const float* __restrict__ We, const float* __restrict__ ae, float* wea) {
  int f = threadIdx.x;
  if (f >= FEDGE) return;
  float s = 0.f;
  #pragma unroll 4
  for (int j = 0; j < FH; ++j) s += We[f * FH + j] * ae[j];
  wea[f] = s;
}

// ---------- per-node scores: one wave per node ----------
__global__ void k_scores(const float* __restrict__ g, const float* __restrict__ a_s,
                         const float* __restrict__ a_d, float* s_src, float* s_dst) {
  int node = (int)(((size_t)blockIdx.x * blockDim.x + threadIdx.x) >> 5);
  int lane = threadIdx.x & 31;
  if (node >= GN) return;
  float4 v  = ((const float4*)(g + (size_t)node * FH))[lane];
  float4 as = ((const float4*)a_s)[lane];
  float4 ad = ((const float4*)a_d)[lane];
  float ps = v.x * as.x + v.y * as.y + v.z * as.z + v.w * as.w;
  float pd = v.x * ad.x + v.y * ad.y + v.z * ad.z + v.w * ad.w;
  #pragma unroll
  for (int off = 16; off > 0; off >>= 1) {
    ps += __shfl_down(ps, off, 32);
    pd += __shfl_down(pd, off, 32);
  }
  if (lane == 0) { s_src[node] = ps; s_dst[node] = pd; }
}

// ---------- edge pass 1: alpha + segment max ----------
__global__ void k_edge1(const int* __restrict__ ei, const float* __restrict__ ea,
                        const float* __restrict__ mea, const float* __restrict__ s_src,
                        const float* __restrict__ s_dst, const float* __restrict__ wea,
                        float* alpha, unsigned* m_ord) {
  int e = blockIdx.x * blockDim.x + threadIdx.x;
  if (e >= GE + GN) return;
  int s, d; const float* eav;
  if (e < GE) { s = ei[e]; d = ei[(size_t)GE + e]; eav = ea + (size_t)e * FEDGE; }
  else        { s = e - GE; d = s;                 eav = mea + (size_t)s * FEDGE; }
  float se = 0.f;
  #pragma unroll
  for (int f = 0; f < FEDGE; ++f) se += eav[f] * wea[f];
  float a = s_src[s] + s_dst[d] + se;
  a = (a >= 0.f) ? a : 0.2f * a;
  alpha[e] = a;
  atomicMax(&m_ord[d], f2ord(a));
}

// ---------- edge pass 2: exp + segment sum ----------
__global__ void k_edge2(const int* __restrict__ ei, float* alpha,
                        const unsigned* __restrict__ m_ord, float* den) {
  int e = blockIdx.x * blockDim.x + threadIdx.x;
  if (e >= GE + GN) return;
  int d = (e < GE) ? ei[(size_t)GE + e] : (e - GE);
  float ex = __expf(alpha[e] - ord2f(m_ord[d]));
  alpha[e] = ex;
  atomicAdd(&den[d], ex);
}

// ---------- edge pass 3: weighted scatter-add, one wave per edge ----------
__global__ void k_edge3(const int* __restrict__ ei, const float* __restrict__ ex,
                        const float* __restrict__ den, const float* __restrict__ g,
                        float* h) {
  int e = (int)(((size_t)blockIdx.x * blockDim.x + threadIdx.x) >> 5);
  int lane = threadIdx.x & 31;
  if (e >= GE + GN) return;
  int s, d;
  if (e < GE) { s = ei[e]; d = ei[(size_t)GE + e]; } else { s = e - GE; d = s; }
  float w = ex[e] / den[d];
  float4 v = ((const float4*)(g + (size_t)s * FH))[lane];
  float* hp = h + (size_t)d * FH + lane * 4;
  atomicAdd(hp + 0, w * v.x);
  atomicAdd(hp + 1, w * v.y);
  atomicAdd(hp + 2, w * v.z);
  atomicAdd(hp + 3, w * v.w);
}

// ---------- bias + relu, writing f16 (feeds next GEMM) ----------
__global__ void k_bias_relu_f16(const float* __restrict__ acc, const float* __restrict__ b,
                                _Float16* __restrict__ outp, size_t total) {
  size_t i = (size_t)blockIdx.x * blockDim.x + threadIdx.x;
  if (i < total) outp[i] = (_Float16)fmaxf(acc[i] + b[i & (FH - 1)], 0.f);
}
// ---------- bias + relu in place (f32, feeds the mean) ----------
__global__ void k_bias_relu(float* h, const float* __restrict__ b, size_t total) {
  size_t i = (size_t)blockIdx.x * blockDim.x + threadIdx.x;
  if (i < total) h[i] = fmaxf(h[i] + b[i & (FH - 1)], 0.f);
}

// ---------- column mean partials ----------
__global__ void k_mean_partial(const float* __restrict__ h, float* hg) {
  int f = threadIdx.x;
  int base = blockIdx.x * 256;
  int end = min(base + 256, GN);
  float s = 0.f;
  for (int i = base; i < end; ++i) s += h[(size_t)i * FH + f];
  atomicAdd(&hg[f], s);
}

// ---------- VAE head: single block (256 threads) ----------
__global__ void k_head(const float* __restrict__ hg, const float* __restrict__ eps,
                       const float* __restrict__ mu_W, const float* __restrict__ mu_b,
                       const float* __restrict__ lv_W, const float* __restrict__ lv_b,
                       const float* __restrict__ dec_W, const float* __restrict__ dec_b,
                       const float* __restrict__ aW1, const float* __restrict__ ab1,
                       const float* __restrict__ aW2, const float* __restrict__ ab2,
                       const float* __restrict__ nW1, const float* __restrict__ nb1,
                       const float* __restrict__ nW2, const float* __restrict__ nb2,
                       float* out_mu, float* out_lv, float* row_adj, float* row_nd) {
  __shared__ float hgs[FH], zsh[LATD], hds[FH], t1[FH], t2[FH];
  int t = threadIdx.x;
  if (t < FH) hgs[t] = hg[t] * (1.0f / (float)GN);
  __syncthreads();
  if (t < LATD) {
    float mu = mu_b[t], lv = lv_b[t];
    for (int k = 0; k < FH; ++k) { mu += hgs[k] * mu_W[k * LATD + t]; lv += hgs[k] * lv_W[k * LATD + t]; }
    out_mu[t] = mu; out_lv[t] = lv;
    zsh[t] = mu + eps[t] * __expf(0.5f * lv);
  }
  __syncthreads();
  if (t < FH) {
    float v = dec_b[t];
    for (int l = 0; l < LATD; ++l) v += zsh[l] * dec_W[l * FH + t];
    hds[t] = fmaxf(v, 0.f);
  }
  __syncthreads();
  if (t < FH) {
    float va = ab1[t], vn = nb1[t];
    for (int k = 0; k < FH; ++k) { va += hds[k] * aW1[k * FH + t]; vn += hds[k] * nW1[k * FH + t]; }
    t1[t] = fmaxf(va, 0.f); t2[t] = fmaxf(vn, 0.f);
  }
  __syncthreads();
  if (t < MAXND) {
    float v = ab2[t];
    for (int k = 0; k < FH; ++k) v += t1[k] * aW2[k * MAXND + t];
    row_adj[t] = v;
  }
  if (t < FH) {
    float v = nb2[t];
    for (int k = 0; k < FH; ++k) v += t2[k] * nW2[k * FH + t];
    row_nd[t] = v;
  }
}

// ---------- rank-1 broadcast into d_out: float4 row stores, no div ----------
// block = 128 threads, grid = GN. t<50: adj row (200 f32 = 50 float4); 64<=t<96: node row (32 float4)
__global__ void k_bcast_rows(const float4* __restrict__ radj4, const float4* __restrict__ rnd4,
                             float4* __restrict__ oadj4, float4* __restrict__ ond4) {
  int n = blockIdx.x;
  int t = threadIdx.x;
  if (t < 50)                 oadj4[(size_t)n * 50 + t] = radj4[t];
  else if (t >= 64 && t < 96) ond4[(size_t)n * 32 + (t - 64)] = rnd4[t - 64];
}

// ---------- orchestration ----------
static inline int cdiv(long long a, long long b) { return (int)((a + b - 1) / b); }

extern "C" void kernel_launch(void* const* d_in, const int* in_sizes, int n_in,
                              void* d_out, int out_size, void* d_ws, size_t ws_size,
                              hipStream_t stream) {
  (void)in_sizes; (void)n_in; (void)out_size; (void)ws_size;
  const float* x    = (const float*)d_in[0];
  const int*   ei   = (const int*)d_in[1];
  const float* ea   = (const float*)d_in[2];
  const float* eps  = (const float*)d_in[3];
  const float* W1   = (const float*)d_in[5];
  const float* as1  = (const float*)d_in[6];
  const float* ad1  = (const float*)d_in[7];
  const float* We1  = (const float*)d_in[8];
  const float* ae1  = (const float*)d_in[9];
  const float* b1   = (const float*)d_in[10];
  const float* W2   = (const float*)d_in[11];
  const float* as2  = (const float*)d_in[12];
  const float* ad2  = (const float*)d_in[13];
  const float* We2  = (const float*)d_in[14];
  const float* ae2  = (const float*)d_in[15];
  const float* b2   = (const float*)d_in[16];
  const float* mu_W = (const float*)d_in[17];
  const float* mu_b = (const float*)d_in[18];
  const float* lv_W = (const float*)d_in[19];
  const float* lv_b = (const float*)d_in[20];
  const float* decW = (const float*)d_in[21];
  const float* decb = (const float*)d_in[22];
  const float* aW1  = (const float*)d_in[23];
  const float* ab1  = (const float*)d_in[24];
  const float* aW2  = (const float*)d_in[25];
  const float* ab2  = (const float*)d_in[26];
  const float* nW1  = (const float*)d_in[27];
  const float* nb1  = (const float*)d_in[28];
  const float* nW2  = (const float*)d_in[29];
  const float* nb2  = (const float*)d_in[30];
  float* out = (float*)d_out;

  // workspace layout (float units; every block 64B-aligned)
  float* W = (float*)d_ws;
  size_t o = 0;
  float*    g     = W + o; o += (size_t)GN * FH;        // f32 GEMM output
  float*    accb  = W + o; o += (size_t)GN * FH;        // f32 aggregation accumulator
  _Float16* x16   = (_Float16*)(W + o); o += (size_t)GN * FH / 2;
  _Float16* h16   = (_Float16*)(W + o); o += (size_t)GN * FH / 2;
  _Float16* Bs    = (_Float16*)(W + o); o += 4 * 8 * 32 * 16 / 2;
  float*    mea   = W + o; o += (size_t)GN * FEDGE;
  float*    alpha = W + o; o += (size_t)(GE + GN);
  float*    s_src = W + o; o += GN;
  float*    s_dst = W + o; o += GN;
  float*    den   = W + o; o += GN;
  float*    deg   = W + o; o += GN;
  unsigned* m_ord = (unsigned*)(W + o); o += GN;
  float*    wea   = W + o; o += 16;
  float*    hg    = W + o; o += 128;
  float*    radj  = W + o; o += 256;
  float*    rnd   = W + o; o += 128;

  const size_t out_mu_off = (size_t)GN * MAXND + (size_t)GN * FH;
  const size_t out_lv_off = out_mu_off + LATD;
  const int T = 256;

  // one-time input conversion + mean edge attr per dst
  k_cvt_f16<<<cdiv((long long)GN * FH / 4, T), T, 0, stream>>>(x, x16, (size_t)GN * FH);
  k_fill_u32<<<cdiv(GN, T), T, 0, stream>>>((unsigned*)deg, 0u, GN);
  k_fill_u32<<<cdiv((long long)GN * FEDGE, T), T, 0, stream>>>((unsigned*)mea, 0u, (size_t)GN * FEDGE);
  k_deg_ea<<<cdiv((long long)GE * 16, T), T, 0, stream>>>(ei, ea, deg, mea);
  k_mea_div<<<cdiv((long long)GN * FEDGE, T), T, 0, stream>>>(deg, mea);

  const _Float16* layer_in[2] = { x16, h16 };
  const float*    layer_W[2]  = { W1, W2 };
  const float*    layer_as[2] = { as1, as2 };
  const float*    layer_ad[2] = { ad1, ad2 };
  const float*    layer_We[2] = { We1, We2 };
  const float*    layer_ae[2] = { ae1, ae2 };
  const float*    layer_b[2]  = { b1, b2 };

  for (int L = 0; L < 2; ++L) {
    k_prep_B<<<4, T, 0, stream>>>(layer_W[L], Bs);
    k_wea<<<1, 32, 0, stream>>>(layer_We[L], layer_ae[L], wea);
    k_gemm128_wmma<<<GN / 16, T, 0, stream>>>(layer_in[L], Bs, g, GN);
    k_scores<<<cdiv((long long)GN * 32, T), T, 0, stream>>>(g, layer_as[L], layer_ad[L], s_src, s_dst);
    k_fill_u32<<<cdiv(GN, T), T, 0, stream>>>(m_ord, 0u, GN);
    k_fill_u32<<<cdiv(GN, T), T, 0, stream>>>((unsigned*)den, 0u, GN);
    k_fill_u32<<<cdiv((long long)GN * FH, T), T, 0, stream>>>((unsigned*)accb, 0u, (size_t)GN * FH);
    k_edge1<<<cdiv(GE + GN, T), T, 0, stream>>>(ei, ea, mea, s_src, s_dst, wea, alpha, m_ord);
    k_edge2<<<cdiv(GE + GN, T), T, 0, stream>>>(ei, alpha, m_ord, den);
    k_edge3<<<cdiv((long long)(GE + GN) * 32, T), T, 0, stream>>>(ei, alpha, den, g, accb);
    if (L == 0)
      k_bias_relu_f16<<<cdiv((long long)GN * FH, T), T, 0, stream>>>(accb, layer_b[L], h16, (size_t)GN * FH);
    else
      k_bias_relu<<<cdiv((long long)GN * FH, T), T, 0, stream>>>(accb, layer_b[L], (size_t)GN * FH);
  }

  k_fill_u32<<<1, 128, 0, stream>>>((unsigned*)hg, 0u, 128);
  k_mean_partial<<<cdiv(GN, 256), 128, 0, stream>>>(accb, hg);
  k_head<<<1, 256, 0, stream>>>(hg, eps, mu_W, mu_b, lv_W, lv_b, decW, decb,
                                aW1, ab1, aW2, ab2, nW1, nb1, nW2, nb2,
                                out + out_mu_off, out + out_lv_off, radj, rnd);
  k_bcast_rows<<<GN, 128, 0, stream>>>((const float4*)radj, (const float4*)rnd,
                                       (float4*)out, (float4*)(out + (size_t)GN * MAXND));
}
